// LinkPredictorGNN_22376779612382
// MI455X (gfx1250) — matrix operand order repaired and checked by reading, not compiled, not run on hardware
//
#include <hip/hip_runtime.h>
#include <hip/hip_bf16.h>
#include <stdint.h>

typedef __bf16 bf16_t;
typedef __attribute__((ext_vector_type(16))) __bf16 v16bf;
typedef __attribute__((ext_vector_type(8)))  float  v8f;
typedef __attribute__((address_space(3)))    bf16_t lds_bf16;

#define NNODES 50000
#define DH     128
#define NEDGES 800000
#define NPAIRS 1000000

static __device__ __forceinline__ v8f zero_v8f() {
    v8f z = {0.f, 0.f, 0.f, 0.f, 0.f, 0.f, 0.f, 0.f};
    return z;
}

// ---------------------------------------------------------------------------
// K0a: pack W_gcn^T to bf16: Wgt[n*128 + k] = W_gcn[k*128 + n]   (128x128)
// ---------------------------------------------------------------------------
__global__ void pack_wgcn(const float* __restrict__ Wg, bf16_t* __restrict__ Wgt) {
    int i = blockIdx.x * blockDim.x + threadIdx.x;
    if (i < 128 * 128) {
        int n = i >> 7, k = i & 127;
        Wgt[n * 128 + k] = (bf16_t)Wg[k * 128 + n];
    }
}

// ---------------------------------------------------------------------------
// K0b: pack W1 into the exact WMMA B-fragment register image:
// W1p layout = [kc(8)][nt(8)][half(2)][lane(32)][e(8)] bf16  (64 KB total).
// For index (kc,nt,half,lane,e):  n = nt*16 + (lane&15)
//                                 k = kc*32 + half*16 + (lane>>4)*8 + e
// value = W1[k*128 + n].  Blocks DMA this linearly into LDS (async-to-LDS)
// and each B fragment is then two conflict-free ds_load_b128 at lane*16.
// ---------------------------------------------------------------------------
__global__ void pack_w1(const float* __restrict__ W1, bf16_t* __restrict__ W1p) {
    int i = blockIdx.x * blockDim.x + threadIdx.x;
    if (i >= 8 * 8 * 2 * 32 * 8) return;
    int kc   = i >> 12;
    int nt   = (i >> 9) & 7;
    int half = (i >> 8) & 1;
    int lane = (i >> 3) & 31;
    int e    = i & 7;
    int n = nt * 16 + (lane & 15);
    int k = kc * 32 + half * 16 + (lane >> 4) * 8 + e;
    W1p[i] = (bf16_t)W1[k * 128 + n];
}

// ---------------------------------------------------------------------------
// Degree: deg[i] = 1 (self loop) + in-degree; then dinv = rsqrt(deg) in place
// ---------------------------------------------------------------------------
__global__ void deg_init(float* __restrict__ deg) {
    int i = blockIdx.x * blockDim.x + threadIdx.x;
    if (i < NNODES) deg[i] = 1.0f;
}
__global__ void deg_count(const long long* __restrict__ ei, float* __restrict__ deg) {
    int e = blockIdx.x * blockDim.x + threadIdx.x;
    if (e < NEDGES) atomicAdd(&deg[(int)ei[NEDGES + e]], 1.0f);
}
__global__ void deg_rsqrt(float* __restrict__ deg) {
    int i = blockIdx.x * blockDim.x + threadIdx.x;
    if (i < NNODES) deg[i] = rsqrtf(deg[i]);
}

// ---------------------------------------------------------------------------
// K4: h0 = x @ W_gcn via bf16 WMMA; write h0s = h0*dinv[row] (edge-scaled copy)
// and h_out init = b_gcn + dinv^2 * h0 (self-loop term). One wave = 16 rows.
// 50000/16 = 3125 tiles exactly -> every WMMA wave has full EXEC.
// ---------------------------------------------------------------------------
__global__ __launch_bounds__(128) void gcn_gemm(
        const float* __restrict__ x, const bf16_t* __restrict__ Wgt,
        const float* __restrict__ dinv, const float* __restrict__ bgcn,
        float* __restrict__ h0s, float* __restrict__ hout) {
    int wid  = blockIdx.x * (blockDim.x >> 5) + (threadIdx.x >> 5);
    if (wid >= NNODES / 16) return;
    int lane = threadIdx.x & 31;
    int hi   = lane >> 4;
    int lm   = lane & 15;
    int m0   = wid * 16;

    v8f acc[8];
#pragma unroll
    for (int nt = 0; nt < 8; nt++) acc[nt] = zero_v8f();

    const float* arow = x + (size_t)(m0 + lm) * DH;
#pragma unroll
    for (int kc = 0; kc < 4; kc++) {
        // A fragment: elements 0..7 -> k = kc*32 + 8*hi + e ; 8..15 -> +16
        const float* c0 = arow + kc * 32 + 8 * hi;
        float4 f0 = *(const float4*)(c0);
        float4 f1 = *(const float4*)(c0 + 4);
        float4 f2 = *(const float4*)(c0 + 16);
        float4 f3 = *(const float4*)(c0 + 20);
        v16bf a;
        a[0]  = (bf16_t)f0.x; a[1]  = (bf16_t)f0.y; a[2]  = (bf16_t)f0.z; a[3]  = (bf16_t)f0.w;
        a[4]  = (bf16_t)f1.x; a[5]  = (bf16_t)f1.y; a[6]  = (bf16_t)f1.z; a[7]  = (bf16_t)f1.w;
        a[8]  = (bf16_t)f2.x; a[9]  = (bf16_t)f2.y; a[10] = (bf16_t)f2.z; a[11] = (bf16_t)f2.w;
        a[12] = (bf16_t)f3.x; a[13] = (bf16_t)f3.y; a[14] = (bf16_t)f3.z; a[15] = (bf16_t)f3.w;
#pragma unroll
        for (int nt = 0; nt < 8; nt++) {
            const bf16_t* brow = Wgt + (nt * 16 + lm) * 128 + kc * 32 + 8 * hi;
            union { v16bf v; uint4 q[2]; } b;
            b.q[0] = *(const uint4*)(brow);
            b.q[1] = *(const uint4*)(brow + 16);
            acc[nt] = __builtin_amdgcn_wmma_f32_16x16x32_bf16(
                false, a, false, b.v, (short)0, acc[nt], false, false);
        }
    }
#pragma unroll
    for (int nt = 0; nt < 8; nt++) {
        int n   = nt * 16 + lm;
        float bc = bgcn[n];
#pragma unroll
        for (int j = 0; j < 8; j++) {
            int   row = m0 + j + 8 * hi;
            float di  = dinv[row];
            float v   = acc[nt][j];
            h0s [(size_t)row * DH + n] = v * di;
            hout[(size_t)row * DH + n] = bc + v * di * di;
        }
    }
}

// ---------------------------------------------------------------------------
// K5: edge scatter — one wave per edge, lane handles 4 columns.
// hout[dst] += h0s[src] * dinv[dst]  (h0s already carries dinv[src])
// ---------------------------------------------------------------------------
__global__ __launch_bounds__(256) void scatter_edges(
        const long long* __restrict__ ei, const float* __restrict__ h0s,
        const float* __restrict__ dinv, float* __restrict__ hout) {
    int wid = blockIdx.x * (blockDim.x >> 5) + (threadIdx.x >> 5);
    if (wid >= NEDGES) return;
    int lane = threadIdx.x & 31;
    int s = (int)ei[wid];
    int d = (int)ei[NEDGES + wid];
    float nd = dinv[d];
    const float* src = h0s + (size_t)s * DH + lane * 4;
    float*       dst = hout + (size_t)d * DH + lane * 4;
    float4 v = *(const float4*)src;
    atomicAdd(dst + 0, v.x * nd);
    atomicAdd(dst + 1, v.y * nd);
    atomicAdd(dst + 2, v.z * nd);
    atomicAdd(dst + 3, v.w * nd);
}

// ---------------------------------------------------------------------------
// K6: h (fp32, final) -> bf16 copy for the edge-MLP GEMM; pairs passthrough
// ---------------------------------------------------------------------------
__global__ void h_to_bf16(const float* __restrict__ h, bf16_t* __restrict__ hb) {
    int i = blockIdx.x * blockDim.x + threadIdx.x;
    if (i < NNODES * DH) hb[i] = (bf16_t)h[i];
}
__global__ void copy_pairs(const long long* __restrict__ p, long long* __restrict__ o) {
    int i = blockIdx.x * blockDim.x + threadIdx.x;
    if (i < 2 * NPAIRS) o[i] = p[i];
}

// ---------------------------------------------------------------------------
// K7: edge MLP. One wave = 32 pairs (two 16-row M-tiles sharing each B
// fragment). W1 is DMA'd memory->LDS with GLOBAL_LOAD_ASYNC_TO_LDS_B128
// (ASYNCcnt-tracked, no VGPR staging), then each B fragment is two
// conflict-free ds_load_b128. K=256 in 8 chunks of 32, N=128 in 8 tiles.
// Layer-2 dot product + bias + ReLU fused in epilogue (shfl_xor reduction).
// 1e6/32 = 31250 wave-tiles exactly -> full EXEC for every WMMA wave.
// ---------------------------------------------------------------------------
__global__ __launch_bounds__(256) void edge_mlp(
        const long long* __restrict__ pairs, const bf16_t* __restrict__ hb,
        const bf16_t* __restrict__ W1p, const float* __restrict__ b1,
        const float* __restrict__ W2, const float* __restrict__ b2,
        float* __restrict__ logits) {
    __shared__ bf16_t sW1[8 * 8 * 512];   // 64 KB, fragment-ordered

    // async DMA fill: 256 threads x 16 x 16B = 64 KB, linear & coalesced.
    // Each lane supplies LDS byte address (VDST) + 64-bit global addr (VADDR).
    {
        unsigned lbase = (unsigned)(uintptr_t)(lds_bf16*)sW1;
#pragma unroll
        for (int i = 0; i < 16; i++) {
            unsigned    idx   = threadIdx.x + i * 256;
            unsigned    laddr = lbase + idx * 16;
            const char* gaddr = (const char*)W1p + (size_t)idx * 16;
            asm volatile("global_load_async_to_lds_b128 %0, %1, off"
                         :: "v"(laddr), "v"(gaddr) : "memory");
        }
        // wait for this wave's async transfers, then barrier for all waves
        asm volatile("s_wait_asynccnt 0" ::: "memory");
    }
    __syncthreads();

    int wid  = blockIdx.x * (blockDim.x >> 5) + (threadIdx.x >> 5);
    if (wid >= NPAIRS / 32) return;           // after barrier; wave-uniform
    int lane = threadIdx.x & 31;
    int hi   = lane >> 4;
    int lm   = lane & 15;
    int p0   = wid * 32;

    // A-row gathers for both M-tiles handled by this lane
    long long u0 = pairs[2 * (p0 + lm) + 0];
    long long v0 = pairs[2 * (p0 + lm) + 1];
    long long u1 = pairs[2 * (p0 + 16 + lm) + 0];
    long long v1 = pairs[2 * (p0 + 16 + lm) + 1];
    const bf16_t* ru0 = hb + (size_t)u0 * DH;
    const bf16_t* rv0 = hb + (size_t)v0 * DH;
    const bf16_t* ru1 = hb + (size_t)u1 * DH;
    const bf16_t* rv1 = hb + (size_t)v1 * DH;

    v8f acc0[8], acc1[8];
#pragma unroll
    for (int nt = 0; nt < 8; nt++) { acc0[nt] = zero_v8f(); acc1[nt] = zero_v8f(); }

#pragma unroll
    for (int kc = 0; kc < 8; kc++) {
        const bf16_t* row0 = (kc < 4) ? (ru0 + kc * 32) : (rv0 + (kc - 4) * 32);
        const bf16_t* row1 = (kc < 4) ? (ru1 + kc * 32) : (rv1 + (kc - 4) * 32);
        union { v16bf v; uint4 q[2]; } a0, a1;
        a0.q[0] = *(const uint4*)(row0 + 8 * hi);
        a0.q[1] = *(const uint4*)(row0 + 16 + 8 * hi);
        a1.q[0] = *(const uint4*)(row1 + 8 * hi);
        a1.q[1] = *(const uint4*)(row1 + 16 + 8 * hi);
#pragma unroll
        for (int nt = 0; nt < 8; nt++) {
            const bf16_t* frag = sW1 + ((kc * 8 + nt) << 9);  // 512 bf16 per fragment
            union { v16bf v; uint4 q[2]; } b;
            b.q[0] = *(const uint4*)(frag + lane * 8);         // half 0: lane*16 bytes
            b.q[1] = *(const uint4*)(frag + 256 + lane * 8);   // half 1
            acc0[nt] = __builtin_amdgcn_wmma_f32_16x16x32_bf16(
                false, a0.v, false, b.v, (short)0, acc0[nt], false, false);
            acc1[nt] = __builtin_amdgcn_wmma_f32_16x16x32_bf16(
                false, a1.v, false, b.v, (short)0, acc1[nt], false, false);
        }
    }

    // Fused layer 2: logit[m] = sum_n relu(acc + b1[n]) * W2[n] + b2
    float pt0[8], pt1[8];
#pragma unroll
    for (int j = 0; j < 8; j++) { pt0[j] = 0.f; pt1[j] = 0.f; }
#pragma unroll
    for (int nt = 0; nt < 8; nt++) {
        int n = nt * 16 + lm;
        float bb = b1[n];
        float ww = W2[n];
#pragma unroll
        for (int j = 0; j < 8; j++) {
            float t0 = acc0[nt][j] + bb;
            float t1 = acc1[nt][j] + bb;
            pt0[j] += (t0 > 0.f ? t0 : 0.f) * ww;
            pt1[j] += (t1 > 0.f ? t1 : 0.f) * ww;
        }
    }
    // reduce over the 16 lanes of this half-wave (xor 1,2,4,8 keeps bit 4)
#pragma unroll
    for (int j = 0; j < 8; j++) {
#pragma unroll
        for (int off = 1; off < 16; off <<= 1) {
            pt0[j] += __shfl_xor(pt0[j], off, 32);
            pt1[j] += __shfl_xor(pt1[j], off, 32);
        }
    }
    float r0 = 0.f, r1 = 0.f;
#pragma unroll
    for (int j = 0; j < 8; j++) {
        if (lm == j) { r0 = pt0[j]; r1 = pt1[j]; }
    }
    if (lm < 8) {
        float bias = b2[0];
        logits[p0 + hi * 8 + lm]      = r0 + bias;   // M-tile 0: rows m = lm + 8*hi
        logits[p0 + 16 + hi * 8 + lm] = r1 + bias;   // M-tile 1
    }
}

// ---------------------------------------------------------------------------
extern "C" void kernel_launch(void* const* d_in, const int* in_sizes, int n_in,
                              void* d_out, int out_size, void* d_ws, size_t ws_size,
                              hipStream_t stream) {
    (void)in_sizes; (void)n_in; (void)out_size; (void)ws_size;
    const float*     x          = (const float*)d_in[0];
    const long long* edge_index = (const long long*)d_in[1];
    const long long* edge_pairs = (const long long*)d_in[2];
    const float*     W_gcn      = (const float*)d_in[3];
    const float*     b_gcn      = (const float*)d_in[4];
    const float*     W1         = (const float*)d_in[5];
    const float*     b1         = (const float*)d_in[6];
    const float*     W2         = (const float*)d_in[7];
    const float*     b2         = (const float*)d_in[8];

    // outputs: [logits: P f32][edge_pairs: 2P i64][h: N*128 f32]
    float*     logits    = (float*)d_out;
    long long* pairs_out = (long long*)(logits + NPAIRS);
    float*     h_out     = (float*)(pairs_out + 2 * NPAIRS);

    // workspace carve-up (all 256B aligned)
    char* ws = (char*)d_ws;
    float*  dinv = (float*)ws;  ws += ((size_t)NNODES * 4 + 255) & ~(size_t)255;
    float*  h0s  = (float*)ws;  ws += (size_t)NNODES * DH * 4;
    bf16_t* hb   = (bf16_t*)ws; ws += (size_t)NNODES * DH * 2;
    bf16_t* Wgt  = (bf16_t*)ws; ws += (size_t)128 * 128 * 2;
    bf16_t* W1p  = (bf16_t*)ws; ws += (size_t)8 * 8 * 512 * 2;

    pack_wgcn<<<(128 * 128 + 255) / 256, 256, 0, stream>>>(W_gcn, Wgt);
    pack_w1  <<<(8 * 8 * 512 + 255) / 256, 256, 0, stream>>>(W1, W1p);
    deg_init <<<(NNODES + 255) / 256, 256, 0, stream>>>(dinv);
    deg_count<<<(NEDGES + 255) / 256, 256, 0, stream>>>(edge_index, dinv);
    deg_rsqrt<<<(NNODES + 255) / 256, 256, 0, stream>>>(dinv);

    // 3125 wave-tiles, 4 waves per 128-thread block
    gcn_gemm<<<(NNODES / 16 + 3) / 4, 128, 0, stream>>>(x, Wgt, dinv, b_gcn, h0s, h_out);

    // 800k edges, 8 waves per 256-thread block
    scatter_edges<<<NEDGES / 8, 256, 0, stream>>>(edge_index, h0s, dinv, h_out);

    h_to_bf16 <<<(NNODES * DH + 255) / 256, 256, 0, stream>>>(h_out, hb);
    copy_pairs<<<(2 * NPAIRS + 255) / 256, 256, 0, stream>>>(edge_pairs, pairs_out);

    // 31250 wave-tiles of 32 pairs, 8 waves per 256-thread block
    edge_mlp<<<(NPAIRS / 32 + 7) / 8, 256, 0, stream>>>(edge_pairs, hb, W1p, b1, W2, b2, logits);
}
